// CorefModel_5583457485602
// MI455X (gfx1250) — matrix operand order repaired and checked by reading, not compiled, not run on hardware
//
#include <hip/hip_runtime.h>
#include <math.h>

// ---------------- model constants ----------------
#define S_   8
#define L_   64
#define NTOK 512          // S*L
#define MW_  10
#define NCAND 5120        // NTOK*MW
#define HID_ 200
#define G4_  800          // 4*HID
#define INP_ 500
#define KTOP 204
#define KPAD 208
#define MAXANT 128
#define NPAIR 26112       // KTOP*MAXANT
#define EMB_ 1320
#define EKP  1344         // EMB padded to mult of 32
#define N160 160
#define NEGV (-1e9f)

typedef _Float16 half16 __attribute__((ext_vector_type(16)));
typedef _Float16 half8  __attribute__((ext_vector_type(8)));
typedef float    floatx8 __attribute__((ext_vector_type(8)));

// ---------------- WMMA fragment helpers (ISA 7.12.2 layouts, wave32) ------
// A fragment: lane&15 selects row; lanes>=16 hold K+8/K+24 halves.
// Loads are two aligned 16B chunks (lda and kb are multiples of 8 halves).
__device__ inline half16 load_a_row(const _Float16* A, int lda, int row, int kb) {
  const half8* p = (const half8*)(A + (size_t)row * lda + kb);
  half8 lo = p[0];   // K = kb+0..7
  half8 hi = p[2];   // K = kb+16..23
  return __builtin_shufflevector(lo, hi, 0, 1, 2, 3, 4, 5, 6, 7,
                                 8, 9, 10, 11, 12, 13, 14, 15);
}

// ---------------- generic GEMM, packed B, 4 M-tiles per wave --------------
// Bp layout: fragment index f = (nt*Kt + kt)*32 + lane, 16 contiguous halves.
__global__ void k_gemm_f16_f32out(const _Float16* __restrict__ A, int lda,
                                  const _Float16* __restrict__ Bp,
                                  const float* __restrict__ bias,
                                  float* __restrict__ C, int ldc,
                                  int Mtiles, int Ntiles, int Kpad, int Nreal) {
  int wave = threadIdx.x >> 5;
  int lane = threadIdx.x & 31;
  int Mgrp = (Mtiles + 3) >> 2;
  int tile = blockIdx.x * 4 + wave;
  if (tile >= Mgrp * Ntiles) return;
  int mg = tile / Ntiles;
  int nt = tile % Ntiles;
  int Kt = Kpad >> 5;
  int mvalid = Mtiles - mg * 4;  // 1..4 valid M-tiles (wave-uniform)
  const half16* bptr = (const half16*)Bp + (size_t)nt * Kt * 32 + lane;
  int mrow = mg * 64 + (lane & 15);
  int kbA = (lane & 16) ? 8 : 0;
  floatx8 a0 = {}, a1 = {}, a2 = {}, a3 = {};
  for (int kt = 0; kt < Kt; ++kt) {
    half16 b = bptr[(size_t)kt * 32];
    int kb = kt * 32 + kbA;
    a0 = __builtin_amdgcn_wmma_f32_16x16x32_f16(false, load_a_row(A, lda, mrow, kb),
                                                false, b, (short)0, a0, false, false);
    if (mvalid > 1)
      a1 = __builtin_amdgcn_wmma_f32_16x16x32_f16(false, load_a_row(A, lda, mrow + 16, kb),
                                                  false, b, (short)0, a1, false, false);
    if (mvalid > 2)
      a2 = __builtin_amdgcn_wmma_f32_16x16x32_f16(false, load_a_row(A, lda, mrow + 32, kb),
                                                  false, b, (short)0, a2, false, false);
    if (mvalid > 3)
      a3 = __builtin_amdgcn_wmma_f32_16x16x32_f16(false, load_a_row(A, lda, mrow + 48, kb),
                                                  false, b, (short)0, a3, false, false);
  }
  int n  = nt * 16 + (lane & 15);
  float bs = (bias && n < Nreal) ? bias[n] : 0.f;
  int mb = mg * 64 + ((lane & 16) ? 8 : 0);
  floatx8 accs[4] = {a0, a1, a2, a3};
  for (int mt = 0; mt < mvalid; ++mt) {
#pragma unroll
    for (int e = 0; e < 8; ++e) {
      int r = mb + mt * 16 + e;
      C[(size_t)r * ldc + n] = (n < Nreal) ? (accs[mt][e] + bs) : 0.f;
    }
  }
}

__global__ void k_gemm_f16_f16out_relu(const _Float16* __restrict__ A, int lda,
                                       const _Float16* __restrict__ Bp,
                                       const float* __restrict__ bias,
                                       _Float16* __restrict__ C, int ldc,
                                       int Mtiles, int Ntiles, int Kpad, int Nreal) {
  int wave = threadIdx.x >> 5;
  int lane = threadIdx.x & 31;
  int Mgrp = (Mtiles + 3) >> 2;
  int tile = blockIdx.x * 4 + wave;
  if (tile >= Mgrp * Ntiles) return;
  int mg = tile / Ntiles;
  int nt = tile % Ntiles;
  int Kt = Kpad >> 5;
  int mvalid = Mtiles - mg * 4;
  const half16* bptr = (const half16*)Bp + (size_t)nt * Kt * 32 + lane;
  int mrow = mg * 64 + (lane & 15);
  int kbA = (lane & 16) ? 8 : 0;
  floatx8 a0 = {}, a1 = {}, a2 = {}, a3 = {};
  for (int kt = 0; kt < Kt; ++kt) {
    half16 b = bptr[(size_t)kt * 32];
    int kb = kt * 32 + kbA;
    a0 = __builtin_amdgcn_wmma_f32_16x16x32_f16(false, load_a_row(A, lda, mrow, kb),
                                                false, b, (short)0, a0, false, false);
    if (mvalid > 1)
      a1 = __builtin_amdgcn_wmma_f32_16x16x32_f16(false, load_a_row(A, lda, mrow + 16, kb),
                                                  false, b, (short)0, a1, false, false);
    if (mvalid > 2)
      a2 = __builtin_amdgcn_wmma_f32_16x16x32_f16(false, load_a_row(A, lda, mrow + 32, kb),
                                                  false, b, (short)0, a2, false, false);
    if (mvalid > 3)
      a3 = __builtin_amdgcn_wmma_f32_16x16x32_f16(false, load_a_row(A, lda, mrow + 48, kb),
                                                  false, b, (short)0, a3, false, false);
  }
  int n  = nt * 16 + (lane & 15);
  float bs = (bias && n < Nreal) ? bias[n] : 0.f;
  int mb = mg * 64 + ((lane & 16) ? 8 : 0);
  floatx8 accs[4] = {a0, a1, a2, a3};
  for (int mt = 0; mt < mvalid; ++mt) {
#pragma unroll
    for (int e = 0; e < 8; ++e) {
      int r = mb + mt * 16 + e;
      float v = (n < Nreal) ? fmaxf(accs[mt][e] + bs, 0.f) : 0.f;
      C[(size_t)r * ldc + n] = (_Float16)v;
    }
  }
}

// ---------------- char CNN + text_emb build -------------------------------
__global__ void k_charcnn(const float* __restrict__ word_emb,
                          const int* __restrict__ char_index,
                          const float* __restrict__ ctab,
                          const float* __restrict__ w3, const float* __restrict__ b3,
                          const float* __restrict__ w4, const float* __restrict__ b4,
                          const float* __restrict__ w5, const float* __restrict__ b5,
                          float* __restrict__ text_emb) {
  __shared__ float ce[16][8];
  int word = blockIdx.x;
  int tid = threadIdx.x;  // 160 threads
  if (tid < 128) {
    int pos = tid >> 3, c = tid & 7;
    ce[pos][c] = ctab[char_index[word * 16 + pos] * 8 + c];
  }
  __syncthreads();
  if (tid < 150) {
    int cv = tid / 50, fi = tid % 50;
    const float* w  = (cv == 0) ? w3 : (cv == 1) ? w4 : w5;
    const float* bb = (cv == 0) ? b3 : (cv == 1) ? b4 : b5;
    int wl = 3 + cv;
    int P  = 16 - wl + 1;
    float m = 0.f;  // relu outputs are >=0
    for (int p = 0; p < P; ++p) {
      float acc = bb[fi];
      for (int c = 0; c < 8; ++c)
        for (int h = 0; h < wl; ++h)
          acc += ce[p + h][c] * w[(fi * 8 + c) * wl + h];
      m = fmaxf(m, fmaxf(acc, 0.f));
    }
    text_emb[(size_t)word * INP_ + 350 + tid] = m;
  }
  for (int d = tid; d < 350; d += 160)
    text_emb[(size_t)word * INP_ + d] = word_emb[(size_t)word * 350 + d];
}

// ---------------- repack text_emb for LSTM input GEMM ---------------------
__global__ void k_build_x16(const float* __restrict__ text_emb,
                            _Float16* __restrict__ x16, int backward) {
  int row = blockIdx.x;            // row = t*8 + s
  int t = row >> 3, s = row & 7;
  int tt = backward ? (63 - t) : t;
  const float* src = text_emb + ((size_t)(s * 64 + tt)) * INP_;
  for (int c = threadIdx.x; c < 512; c += 256)
    x16[(size_t)row * 512 + c] = (_Float16)(c < INP_ ? src[c] : 0.f);
}

// ---------------- weight pack: f32 [N][K] -> f16 packed B fragments -------
// Bp fragment f = (nt*(Kpad/32) + kt)*32 + lane holds 16 contiguous halves:
//   element e -> B[kt*32 + ((lane&16)?8:0) + (e&7) + ((e&8)?16:0)][nt*16 + (lane&15)]
__global__ void k_w_pack(const float* __restrict__ W, int wld, int Nreal, int Kreal,
                         int coloff, _Float16* __restrict__ Bp, int Kpad, int Npad) {
  int idx = blockIdx.x * 256 + threadIdx.x;
  if (idx >= Kpad * Npad) return;
  int e    = idx & 15;
  int lane = (idx >> 4) & 31;
  int frag = idx >> 9;
  int Kt = Kpad >> 5;
  int kt = frag % Kt;
  int nt = frag / Kt;
  int k = kt * 32 + ((lane & 16) ? 8 : 0) + (e & 7) + ((e & 8) ? 16 : 0);
  int n = nt * 16 + (lane & 15);
  float v = (k < Kreal && n < Nreal) ? W[(size_t)n * wld + coloff + k] : 0.f;
  Bp[idx] = (_Float16)v;
}

__global__ void k_bias_sum(const float* a, const float* b, float* o) {
  int g = blockIdx.x * 256 + threadIdx.x;
  if (g < G4_) o[g] = a[g] + b[g];
}

// ---------------- LSTM recurrence (one workgroup, LDS-resident state) -----
__device__ inline float sigf(float x) { return 1.f / (1.f + expf(-x)); }

__global__ void k_lstm(const float* __restrict__ gates,   // [64*8][800], row t*8+b
                       const float* __restrict__ Whh,     // [800][200]
                       const float* __restrict__ h0,      // [8][200]
                       const float* __restrict__ c0,
                       float* __restrict__ text_out,      // [512][400]
                       int backward) {
  __shared__ float hs[8][HID_];
  __shared__ float cs[8][HID_];
  __shared__ float gb[8][G4_];
  int tid = threadIdx.x;  // 800
  for (int idx = tid; idx < 8 * HID_; idx += G4_) {
    int b = idx / HID_, g = idx % HID_;
    hs[b][g] = h0[b * HID_ + g];
    cs[b][g] = c0[b * HID_ + g];
  }
  __syncthreads();
  for (int t = 0; t < 64; ++t) {
    const float* wr = Whh + (size_t)tid * HID_;
    for (int b = 0; b < 8; ++b) {
      float acc = gates[((size_t)(t * 8 + b)) * G4_ + tid];
      for (int k = 0; k < HID_; ++k) acc += hs[b][k] * wr[k];
      gb[b][tid] = acc;
    }
    __syncthreads();
    if (tid < HID_) {
      int tt = backward ? (63 - t) : t;
      for (int b = 0; b < 8; ++b) {
        float gi = gb[b][tid], gf = gb[b][HID_ + tid];
        float gg = gb[b][2 * HID_ + tid], go = gb[b][3 * HID_ + tid];
        float c = sigf(gf) * cs[b][tid] + sigf(gi) * tanhf(gg);
        float h = sigf(go) * tanhf(c);
        cs[b][tid] = c;
        hs[b][tid] = h;
        text_out[((size_t)(b * 64 + tt)) * 400 + (backward ? HID_ : 0) + tid] = h;
      }
    }
    __syncthreads();
  }
}

// ---------------- head scores ---------------------------------------------
__global__ void k_head_scores(const float* __restrict__ to, const float* __restrict__ hw,
                              const float* __restrict__ hb, float* __restrict__ hsc) {
  int n = blockIdx.x * blockDim.x + threadIdx.x;
  if (n >= NTOK) return;
  float acc = hb[0];
  for (int k = 0; k < 400; ++k) acc += to[(size_t)n * 400 + k] * hw[k];
  hsc[n] = acc;
}

// ---------------- candidate span embeddings (f16, padded to 1344) ---------
__global__ void k_cand_emb(const float* __restrict__ text_out,
                           const float* __restrict__ text_emb,
                           const float* __restrict__ head_sc,
                           const float* __restrict__ width_emb,
                           _Float16* __restrict__ cand16) {
  __shared__ float sc[MW_];
  __shared__ float at[MW_];
  int c = blockIdx.x;
  int tid = threadIdx.x;  // 128
  int start = c / MW_, wdt = c % MW_;
  int endc = start + wdt; if (endc > NTOK - 1) endc = NTOK - 1;
  if (tid < MW_) {
    int idx = start + tid; if (idx > NTOK - 1) idx = NTOK - 1;
    sc[tid] = (tid <= wdt) ? head_sc[idx] : NEGV;
  }
  __syncthreads();
  if (tid == 0) {
    float mx = sc[0];
    for (int p = 1; p < MW_; ++p) mx = fmaxf(mx, sc[p]);
    float s = 0.f;
    for (int p = 0; p < MW_; ++p) { at[p] = expf(sc[p] - mx); s += at[p]; }
    for (int p = 0; p < MW_; ++p) at[p] /= s;
  }
  __syncthreads();
  _Float16* dst = cand16 + (size_t)c * EKP;
  for (int d = tid; d < EKP; d += 128) {
    float v = 0.f;
    if (d < 400) {
      v = text_out[(size_t)start * 400 + d];
    } else if (d < 800) {
      v = text_out[(size_t)endc * 400 + (d - 400)];
    } else if (d < 1300) {
      int hd = d - 800;
      for (int p = 0; p < MW_; ++p) {
        int idx = start + p; if (idx > NTOK - 1) idx = NTOK - 1;
        v += at[p] * text_emb[(size_t)idx * INP_ + hd];
      }
    } else if (d < EMB_) {
      v = width_emb[wdt * 20 + (d - 1300)];
    }
    dst[d] = (_Float16)v;
  }
}

// ---------------- mention final score -------------------------------------
__global__ void k_mention_score(const _Float16* __restrict__ h1,
                                const float* __restrict__ w2, const float* __restrict__ b2,
                                float* __restrict__ scv) {
  int c = blockIdx.x * 256 + threadIdx.x;
  if (c >= NCAND) return;
  int start = c / MW_, w = c % MW_, end = start + w;
  float s = b2[0];
  const _Float16* hp = h1 + (size_t)c * N160;
  for (int k = 0; k < 150; ++k) s += (float)hp[k] * w2[k];
  s = fmaxf(s, 0.f);
  bool valid = (end < NTOK) && (start / L_ == end / L_);
  scv[c] = valid ? s : NEGV;
}

// ---------------- top-K by rank (exact lax.top_k tie semantics) -----------
__global__ void k_topk_flags(const float* __restrict__ sc, int* __restrict__ flags) {
  int i = blockIdx.x * 256 + threadIdx.x;
  if (i >= NCAND) return;
  float si = sc[i];
  int cnt = 0;
  for (int j = 0; j < NCAND; ++j) {
    float sj = sc[j];
    cnt += (sj > si) || (sj == si && j < i);
  }
  flags[i] = (cnt < KTOP) ? 1 : 0;
}

__global__ void k_compact(const int* __restrict__ flags, const float* __restrict__ sc,
                          const int* __restrict__ spk_ids,
                          const int* __restrict__ gs, const int* __restrict__ ge,
                          const int* __restrict__ cl,
                          int* __restrict__ m_idx, int* __restrict__ m_spk,
                          int* __restrict__ m_cl, float* __restrict__ m_sc) {
  int i = blockIdx.x * 256 + threadIdx.x;
  if (i >= NCAND || !flags[i]) return;
  int pos = 0;
  for (int j = 0; j < i; ++j) pos += flags[j];
  if (pos >= KTOP) return;
  int st = i / MW_, w = i % MW_, en = st + w;
  if (en > NTOK - 1) en = NTOK - 1;
  m_idx[pos] = i;
  m_spk[pos] = spk_ids[st];
  m_sc[pos]  = sc[i];
  int acc = 0;
  for (int g = 0; g < 100; ++g)
    if (st == gs[g] && en == ge[g]) acc += cl[g];
  m_cl[pos] = acc;
}

__global__ void k_gather_memb(const _Float16* __restrict__ cand, const int* __restrict__ m_idx,
                              _Float16* __restrict__ memb) {
  int k = blockIdx.x;  // 0..207
  int tid = threadIdx.x;
  if (k < KTOP) {
    const _Float16* src = cand + (size_t)m_idx[k] * EKP;
    for (int d = tid; d < EKP; d += 256) memb[(size_t)k * EKP + d] = src[d];
  } else {
    for (int d = tid; d < EKP; d += 256) memb[(size_t)k * EKP + d] = (_Float16)0.f;
  }
}

// ---------------- feature tables: spk/genre/dist columns of ant w0 --------
__global__ void k_feat_tables(const float* __restrict__ aw0,  // [150][4020]
                              const float* __restrict__ spk_emb,
                              const float* __restrict__ genre_emb,
                              const float* __restrict__ dist_emb,
                              const int* __restrict__ genre_ptr,
                              float* __restrict__ spk_term, float* __restrict__ genre_term,
                              float* __restrict__ dist_term) {
  int n = threadIdx.x;  // 160
  int g = *genre_ptr;
  if (n < 150) {
    const float* wr = aw0 + (size_t)n * 4020;
    for (int s = 0; s < 2; ++s) {
      float acc = 0.f;
      for (int f = 0; f < 20; ++f) acc += wr[3960 + f] * spk_emb[s * 20 + f];
      spk_term[s * N160 + n] = acc;
    }
    float accg = 0.f;
    for (int f = 0; f < 20; ++f) accg += wr[3980 + f] * genre_emb[g * 20 + f];
    genre_term[n] = accg;
    for (int d = 0; d < 10; ++d) {
      float acc = 0.f;
      for (int f = 0; f < 20; ++f) acc += wr[4000 + f] * dist_emb[d * 20 + f];
      dist_term[d * N160 + n] = acc;
    }
  } else {
    spk_term[n] = 0.f; spk_term[N160 + n] = 0.f;
    genre_term[n] = 0.f;
    for (int d = 0; d < 10; ++d) dist_term[d * N160 + n] = 0.f;
  }
}

// ---------------- fused pair product-term WMMA + layer-0 epilogue ---------
// 4 M-tiles (64 pair rows) per wave; product fragments built with packed
// f16 vector multiplies; B fragment is one packed 32B load per K-step.
__device__ inline half16 load_prod_frag(const _Float16* pi, const _Float16* pj, int kb) {
  const half8* qi = (const half8*)(pi + kb);
  const half8* qj = (const half8*)(pj + kb);
  half8 lo = qi[0] * qj[0];
  half8 hi = qi[2] * qj[2];
  return __builtin_shufflevector(lo, hi, 0, 1, 2, 3, 4, 5, 6, 7,
                                 8, 9, 10, 11, 12, 13, 14, 15);
}

__global__ void k_pair_gemm(const _Float16* __restrict__ memb,   // [208][1344]
                            const _Float16* __restrict__ Bcp,    // packed [1344][160]
                            const float* __restrict__ Ai, const float* __restrict__ Bj,
                            const float* __restrict__ b0,
                            const float* __restrict__ spk_term,
                            const float* __restrict__ genre_term,
                            const float* __restrict__ dist_term,
                            const int* __restrict__ m_spk,
                            _Float16* __restrict__ h0out) {     // [26112][160]
  int wave = threadIdx.x >> 5;
  int lane = threadIdx.x & 31;
  const int Ntiles = N160 / 16;
  const int Mgrp = NPAIR / 64;      // 408
  int tile = blockIdx.x * 4 + wave;
  if (tile >= Mgrp * Ntiles) return;
  int mg = tile / Ntiles;
  int nt = tile % Ntiles;
  const int Kt = EKP / 32;          // 42
  const half16* bptr = (const half16*)Bcp + (size_t)nt * Kt * 32 + lane;
  int kbA = (lane & 16) ? 8 : 0;

  const _Float16* pi[4];
  const _Float16* pj[4];
#pragma unroll
  for (int mt = 0; mt < 4; ++mt) {
    int r = mg * 64 + mt * 16 + (lane & 15);
    int i = r >> 7, a = r & 127;
    int j = i - MAXANT + a; if (j < 0) j = 0;
    pi[mt] = memb + (size_t)i * EKP;
    pj[mt] = memb + (size_t)j * EKP;
  }

  floatx8 a0 = {}, a1 = {}, a2 = {}, a3 = {};
  for (int kt = 0; kt < Kt; ++kt) {
    half16 b = bptr[(size_t)kt * 32];
    int kb = kt * 32 + kbA;
    a0 = __builtin_amdgcn_wmma_f32_16x16x32_f16(false, load_prod_frag(pi[0], pj[0], kb),
                                                false, b, (short)0, a0, false, false);
    a1 = __builtin_amdgcn_wmma_f32_16x16x32_f16(false, load_prod_frag(pi[1], pj[1], kb),
                                                false, b, (short)0, a1, false, false);
    a2 = __builtin_amdgcn_wmma_f32_16x16x32_f16(false, load_prod_frag(pi[2], pj[2], kb),
                                                false, b, (short)0, a2, false, false);
    a3 = __builtin_amdgcn_wmma_f32_16x16x32_f16(false, load_prod_frag(pi[3], pj[3], kb),
                                                false, b, (short)0, a3, false, false);
  }

  int n  = nt * 16 + (lane & 15);
  int mb = mg * 64 + ((lane & 16) ? 8 : 0);
  floatx8 accs[4] = {a0, a1, a2, a3};
  for (int mt = 0; mt < 4; ++mt) {
#pragma unroll
    for (int e = 0; e < 8; ++e) {
      int r = mb + mt * 16 + e;
      int i = r >> 7, a = r & 127;
      int j = i - MAXANT + a;
      int jc = (j < 0) ? 0 : j;
      float v = 0.f;
      if (n < 150) {
        int d = i - jc;
        int bkt = (d < 5) ? d : (31 - __builtin_clz(d)) + 3;
        if (bkt > 9) bkt = 9;
        int same = (m_spk[i] == m_spk[jc]) ? 1 : 0;
        v = accs[mt][e] + Ai[i * N160 + n] + Bj[jc * N160 + n] + b0[n]
          + spk_term[same * N160 + n] + genre_term[n] + dist_term[bkt * N160 + n];
        v = fmaxf(v, 0.f);
      }
      h0out[(size_t)r * N160 + n] = (_Float16)v;
    }
  }
}

// ---------------- pair final score + output assembly ----------------------
__global__ void k_pair_final(const _Float16* __restrict__ h1,
                             const float* __restrict__ w2, const float* __restrict__ b2,
                             const float* __restrict__ m_sc, float* __restrict__ out) {
  int r = blockIdx.x * 256 + threadIdx.x;
  if (r >= NPAIR) return;
  int i = r >> 7, a = r & 127;
  int j = i - MAXANT + a;
  int jc = (j < 0) ? 0 : j;
  const _Float16* hp = h1 + (size_t)r * N160;
  float s = b2[0];
  for (int k = 0; k < 150; ++k) s += (float)hp[k] * w2[k];
  s = fmaxf(s, 0.f);
  float val = s + m_sc[i] + m_sc[jc] + (j >= 0 ? 0.f : NEGV);
  out[i * (MAXANT + 1) + 1 + a] = val;
  if (a == 0) out[i * (MAXANT + 1)] = 0.f;
}

// ---------------- marginalized softmax loss (single block, deterministic) -
__global__ void k_loss(const float* __restrict__ scores, const int* __restrict__ mcl,
                       float* __restrict__ out) {
  __shared__ float red[256];
  int tid = threadIdx.x;
  float lsum = 0.f;
  for (int i = tid; i < KTOP; i += 256) {
    const float* row = scores + (size_t)i * (MAXANT + 1);
    float mx = row[0];
    for (int c = 1; c <= MAXANT; ++c) mx = fmaxf(mx, row[c]);
    float se = 0.f;
    for (int c = 0; c <= MAXANT; ++c) se += expf(row[c] - mx);
    float lse_all = mx + logf(se);
    int ci = mcl[i];
    int any = 0;
    for (int a = 0; a < MAXANT; ++a) {
      int j = i - MAXANT + a;
      if (j >= 0 && ci > 0 && mcl[j] == ci) any = 1;
    }
    float lse_lab;
    if (!any) {
      lse_lab = row[0];  // dummy only (== 0)
    } else {
      float mxl = -3.0e38f;
      for (int a = 0; a < MAXANT; ++a) {
        int j = i - MAXANT + a;
        if (j >= 0 && ci > 0 && mcl[j] == ci) mxl = fmaxf(mxl, row[1 + a]);
      }
      float sl = 0.f;
      for (int a = 0; a < MAXANT; ++a) {
        int j = i - MAXANT + a;
        if (j >= 0 && ci > 0 && mcl[j] == ci) sl += expf(row[1 + a] - mxl);
      }
      lse_lab = mxl + logf(sl);
    }
    lsum += lse_all - lse_lab;
  }
  red[tid] = lsum;
  __syncthreads();
  for (int s = 128; s > 0; s >>= 1) {
    if (tid < s) red[tid] += red[tid + s];
    __syncthreads();
  }
  if (tid == 0) out[KTOP * (MAXANT + 1)] = red[0];
}

// ---------------- host-side orchestration ---------------------------------
enum {
  IN_WORD_EMB = 0, IN_CHAR_IDX, IN_TEXT_LEN, IN_SPEAKER, IN_GENRE,
  IN_GOLD_S, IN_GOLD_E, IN_CLUSTER,
  IN_CHAR_TAB, IN_W3, IN_B3, IN_W4, IN_B4, IN_W5, IN_B5,
  IN_F_WIH, IN_F_WHH, IN_F_BIH, IN_F_BHH, IN_F_H0, IN_F_C0,
  IN_B_WIH, IN_B_WHH, IN_B_BIH, IN_B_BHH, IN_B_H0, IN_B_C0,
  IN_HEAD_W, IN_HEAD_B, IN_WIDTH_EMB, IN_GENRE_EMB, IN_SPK_EMB, IN_DIST_EMB,
  IN_M_W0, IN_M_B0, IN_M_W1, IN_M_B1, IN_M_W2, IN_M_B2,
  IN_A_W0, IN_A_B0, IN_A_W1, IN_A_B1, IN_A_W2, IN_A_B2
};

extern "C" void kernel_launch(void* const* d_in, const int* in_sizes, int n_in,
                              void* d_out, int out_size, void* d_ws, size_t ws_size,
                              hipStream_t stream) {
  (void)in_sizes; (void)n_in; (void)out_size; (void)ws_size;
  const float* word_emb  = (const float*)d_in[IN_WORD_EMB];
  const int*   char_idx  = (const int*)d_in[IN_CHAR_IDX];
  const int*   spk_ids   = (const int*)d_in[IN_SPEAKER];
  const int*   genre_ptr = (const int*)d_in[IN_GENRE];
  const int*   gold_s    = (const int*)d_in[IN_GOLD_S];
  const int*   gold_e    = (const int*)d_in[IN_GOLD_E];
  const int*   cluster   = (const int*)d_in[IN_CLUSTER];
  const float* char_tab  = (const float*)d_in[IN_CHAR_TAB];
  const float* w3 = (const float*)d_in[IN_W3];  const float* b3 = (const float*)d_in[IN_B3];
  const float* w4 = (const float*)d_in[IN_W4];  const float* b4 = (const float*)d_in[IN_B4];
  const float* w5 = (const float*)d_in[IN_W5];  const float* b5 = (const float*)d_in[IN_B5];
  const float* f_wih = (const float*)d_in[IN_F_WIH];
  const float* f_whh = (const float*)d_in[IN_F_WHH];
  const float* f_bih = (const float*)d_in[IN_F_BIH];
  const float* f_bhh = (const float*)d_in[IN_F_BHH];
  const float* f_h0  = (const float*)d_in[IN_F_H0];
  const float* f_c0  = (const float*)d_in[IN_F_C0];
  const float* b_wih = (const float*)d_in[IN_B_WIH];
  const float* b_whh = (const float*)d_in[IN_B_WHH];
  const float* b_bih = (const float*)d_in[IN_B_BIH];
  const float* b_bhh = (const float*)d_in[IN_B_BHH];
  const float* b_h0  = (const float*)d_in[IN_B_H0];
  const float* b_c0  = (const float*)d_in[IN_B_C0];
  const float* head_w = (const float*)d_in[IN_HEAD_W];
  const float* head_b = (const float*)d_in[IN_HEAD_B];
  const float* width_emb = (const float*)d_in[IN_WIDTH_EMB];
  const float* genre_emb = (const float*)d_in[IN_GENRE_EMB];
  const float* spk_emb   = (const float*)d_in[IN_SPK_EMB];
  const float* dist_emb  = (const float*)d_in[IN_DIST_EMB];
  const float* m_w0 = (const float*)d_in[IN_M_W0];  const float* m_b0 = (const float*)d_in[IN_M_B0];
  const float* m_w1 = (const float*)d_in[IN_M_W1];  const float* m_b1 = (const float*)d_in[IN_M_B1];
  const float* m_w2 = (const float*)d_in[IN_M_W2];  const float* m_b2 = (const float*)d_in[IN_M_B2];
  const float* a_w0 = (const float*)d_in[IN_A_W0];  const float* a_b0 = (const float*)d_in[IN_A_B0];
  const float* a_w1 = (const float*)d_in[IN_A_W1];  const float* a_b1 = (const float*)d_in[IN_A_B1];
  const float* a_w2 = (const float*)d_in[IN_A_W2];  const float* a_b2 = (const float*)d_in[IN_A_B2];

  // workspace carve
  char* base = (char*)d_ws;
  size_t off = 0;
  auto alloc = [&](size_t nbytes) -> void* {
    void* p = base + off;
    off += (nbytes + 255) & ~(size_t)255;
    return p;
  };
  float*    text_emb = (float*)alloc((size_t)NTOK * INP_ * 4);
  float*    text_out = (float*)alloc((size_t)NTOK * 400 * 4);
  _Float16* x16f  = (_Float16*)alloc((size_t)512 * 512 * 2);
  _Float16* x16b  = (_Float16*)alloc((size_t)512 * 512 * 2);
  _Float16* wihTf = (_Float16*)alloc((size_t)512 * G4_ * 2);
  _Float16* wihTb = (_Float16*)alloc((size_t)512 * G4_ * 2);
  float*    bsumf = (float*)alloc(G4_ * 4);
  float*    bsumb = (float*)alloc(G4_ * 4);
  float*    gatesf = (float*)alloc((size_t)512 * G4_ * 4);
  float*    gatesb = (float*)alloc((size_t)512 * G4_ * 4);
  float*    head_sc = (float*)alloc(NTOK * 4);
  _Float16* cand16 = (_Float16*)alloc((size_t)NCAND * EKP * 2);
  _Float16* mw0T = (_Float16*)alloc((size_t)EKP * N160 * 2);
  _Float16* mw1T = (_Float16*)alloc((size_t)N160 * N160 * 2);
  _Float16* mh0  = (_Float16*)alloc((size_t)NCAND * N160 * 2);
  _Float16* mh1  = (_Float16*)alloc((size_t)NCAND * N160 * 2);
  float*    cand_sc = (float*)alloc(NCAND * 4);
  int*      flags   = (int*)alloc(NCAND * 4);
  int*      m_idx   = (int*)alloc(256 * 4);
  int*      m_spk   = (int*)alloc(256 * 4);
  int*      m_cl    = (int*)alloc(256 * 4);
  float*    m_sc    = (float*)alloc(256 * 4);
  _Float16* memb  = (_Float16*)alloc((size_t)KPAD * EKP * 2);
  _Float16* aw0aT = (_Float16*)alloc((size_t)EKP * N160 * 2);
  _Float16* aw0bT = (_Float16*)alloc((size_t)EKP * N160 * 2);
  _Float16* aw0cT = (_Float16*)alloc((size_t)EKP * N160 * 2);
  _Float16* aw1T  = (_Float16*)alloc((size_t)N160 * N160 * 2);
  float*    Ai = (float*)alloc((size_t)KPAD * N160 * 4);
  float*    Bj = (float*)alloc((size_t)KPAD * N160 * 4);
  float*    spk_term   = (float*)alloc(2 * N160 * 4);
  float*    genre_term = (float*)alloc(N160 * 4);
  float*    dist_term  = (float*)alloc(10 * N160 * 4);
  _Float16* ph0 = (_Float16*)alloc((size_t)NPAIR * N160 * 2);
  _Float16* ph1 = (_Float16*)alloc((size_t)NPAIR * N160 * 2);
  float* out = (float*)d_out;

  // 1. char CNN + text embedding
  k_charcnn<<<NTOK, 160, 0, stream>>>(word_emb, char_idx, char_tab, w3, b3, w4, b4, w5, b5,
                                      text_emb);
  // 2. LSTM input repack + packed weights + gate GEMMs (WMMA)
  k_build_x16<<<512, 256, 0, stream>>>(text_emb, x16f, 0);
  k_build_x16<<<512, 256, 0, stream>>>(text_emb, x16b, 1);
  k_w_pack<<<(512 * G4_) / 256, 256, 0, stream>>>(f_wih, 500, G4_, 500, 0, wihTf, 512, G4_);
  k_w_pack<<<(512 * G4_) / 256, 256, 0, stream>>>(b_wih, 500, G4_, 500, 0, wihTb, 512, G4_);
  k_bias_sum<<<4, 256, 0, stream>>>(f_bih, f_bhh, bsumf);
  k_bias_sum<<<4, 256, 0, stream>>>(b_bih, b_bhh, bsumb);
  k_gemm_f16_f32out<<<(8 * 50 + 3) / 4, 128, 0, stream>>>(x16f, 512, wihTf, bsumf,
                                                          gatesf, G4_, 32, 50, 512, G4_);
  k_gemm_f16_f32out<<<(8 * 50 + 3) / 4, 128, 0, stream>>>(x16b, 512, wihTb, bsumb,
                                                          gatesb, G4_, 32, 50, 512, G4_);
  // 3. LSTM recurrence (one WGP per direction)
  k_lstm<<<1, G4_, 0, stream>>>(gatesf, f_whh, f_h0, f_c0, text_out, 0);
  k_lstm<<<1, G4_, 0, stream>>>(gatesb, b_whh, b_h0, b_c0, text_out, 1);
  // 4. head scores + candidate embeddings
  k_head_scores<<<2, 256, 0, stream>>>(text_out, head_w, head_b, head_sc);
  k_cand_emb<<<NCAND, 128, 0, stream>>>(text_out, text_emb, head_sc, width_emb, cand16);
  // 5. mention FFNN (WMMA)
  k_w_pack<<<(EKP * N160) / 256, 256, 0, stream>>>(m_w0, 1320, 150, 1320, 0, mw0T, EKP, N160);
  k_w_pack<<<(N160 * N160) / 256, 256, 0, stream>>>(m_w1, 150, 150, 150, 0, mw1T, N160, N160);
  k_gemm_f16_f16out_relu<<<(80 * 10 + 3) / 4, 128, 0, stream>>>(cand16, EKP, mw0T, m_b0,
                                                                mh0, N160, 320, 10, EKP, 150);
  k_gemm_f16_f16out_relu<<<(80 * 10 + 3) / 4, 128, 0, stream>>>(mh0, N160, mw1T, m_b1,
                                                                mh1, N160, 320, 10, N160, 150);
  k_mention_score<<<NCAND / 256, 256, 0, stream>>>(mh1, m_w2, m_b2, cand_sc);
  // 6. top-K prune (rank-based, index-sorted)
  k_topk_flags<<<NCAND / 256, 256, 0, stream>>>(cand_sc, flags);
  k_compact<<<NCAND / 256, 256, 0, stream>>>(flags, cand_sc, spk_ids, gold_s, gold_e, cluster,
                                             m_idx, m_spk, m_cl, m_sc);
  k_gather_memb<<<KPAD, 256, 0, stream>>>(cand16, m_idx, memb);
  // 7. antecedent FFNN layer0 decomposition
  k_w_pack<<<(EKP * N160) / 256, 256, 0, stream>>>(a_w0, 4020, 150, 1320, 0, aw0aT, EKP, N160);
  k_w_pack<<<(EKP * N160) / 256, 256, 0, stream>>>(a_w0, 4020, 150, 1320, 1320, aw0bT, EKP, N160);
  k_w_pack<<<(EKP * N160) / 256, 256, 0, stream>>>(a_w0, 4020, 150, 1320, 2640, aw0cT, EKP, N160);
  k_w_pack<<<(N160 * N160) / 256, 256, 0, stream>>>(a_w1, 150, 150, 150, 0, aw1T, N160, N160);
  k_feat_tables<<<1, 160, 0, stream>>>(a_w0, spk_emb, genre_emb, dist_emb, genre_ptr,
                                       spk_term, genre_term, dist_term);
  k_gemm_f16_f32out<<<(4 * 10 + 3) / 4, 128, 0, stream>>>(memb, EKP, aw0aT, nullptr,
                                                          Ai, N160, 13, 10, EKP, N160);
  k_gemm_f16_f32out<<<(4 * 10 + 3) / 4, 128, 0, stream>>>(memb, EKP, aw0bT, nullptr,
                                                          Bj, N160, 13, 10, EKP, N160);
  // 8. fused pair product WMMA + layer0 epilogue, then layer1 + final
  k_pair_gemm<<<(408 * 10 + 3) / 4, 128, 0, stream>>>(memb, aw0cT, Ai, Bj, a_b0,
                                                      spk_term, genre_term, dist_term,
                                                      m_spk, ph0);
  k_gemm_f16_f16out_relu<<<(408 * 10 + 3) / 4, 128, 0, stream>>>(
      ph0, N160, aw1T, a_b1, ph1, N160, NPAIR / 16, 10, N160, 150);
  k_pair_final<<<NPAIR / 256, 256, 0, stream>>>(ph1, a_w2, a_b2, m_sc, out);
  // 9. loss
  k_loss<<<1, 256, 0, stream>>>(out, m_cl, out);
}
